// GCNNet_61323543052323
// MI455X (gfx1250) — compile-verified, matrix operand order checked by reading, not run on hardware
//
#include <hip/hip_runtime.h>
#include <hip/hip_bf16.h>

#define NN   100000
#define EE   800000
#define FIN  512
#define HID  256
#define DOUT 128

typedef __attribute__((ext_vector_type(16))) __bf16 v16bf;
typedef __attribute__((ext_vector_type(8)))  float  v8f;

__device__ __forceinline__ unsigned short f2bf(float f) {
  unsigned int u = __float_as_uint(f);
  u += 0x7FFFu + ((u >> 16) & 1u);          // round-to-nearest-even
  return (unsigned short)(u >> 16);
}

// ---------------- elementwise conversions ----------------
__global__ void __launch_bounds__(256) cvt_bf16_kernel(const float* __restrict__ in,
                                                       unsigned short* __restrict__ out,
                                                       long long n) {
  long long i = (long long)blockIdx.x * blockDim.x + threadIdx.x;
  if (i < n) out[i] = f2bf(in[i]);
}

// W [K,N] f32 row-major  ->  Wt [N,K] bf16 row-major (B^T for contiguous fragment loads)
__global__ void __launch_bounds__(256) cvt_transpose_kernel(const float* __restrict__ W,
                                                            unsigned short* __restrict__ Wt,
                                                            int K, int N) {
  int i = blockIdx.x * blockDim.x + threadIdx.x;
  if (i < K * N) {
    int n = i / K, k = i % K;
    Wt[i] = f2bf(W[(long long)k * N + n]);
  }
}

// ---------------- degree / normalization ----------------
__global__ void __launch_bounds__(256) deg_init_kernel(float* __restrict__ deg, int n) {
  int i = blockIdx.x * blockDim.x + threadIdx.x;
  if (i < n) deg[i] = 1.0f;                 // self-loop
}
__global__ void __launch_bounds__(256) deg_accum_kernel(const int* __restrict__ ei,
                                                        float* __restrict__ deg, int e) {
  int i = blockIdx.x * blockDim.x + threadIdx.x;
  if (i < e) atomicAdd(&deg[ei[e + i]], 1.0f);   // dst row of edge_index
}
__global__ void __launch_bounds__(256) deg_rsqrt_kernel(float* __restrict__ deg, int n) {
  int i = blockIdx.x * blockDim.x + threadIdx.x;
  if (i < n) deg[i] = rsqrtf(deg[i]);       // deg >= 1 always
}

// ---------------- WMMA bf16 GEMM: C[M,N] = A[M,K] @ Bt[N,K]^T ----------------
// wave tile: 32(M) x 64(N); K step 32; M%32==0, N%64==0, K%32==0
__global__ void __launch_bounds__(128) wmma_gemm_bf16(const unsigned short* __restrict__ A,
                                                      const unsigned short* __restrict__ Bt,
                                                      float* __restrict__ C,
                                                      int M, int N, int K) {
  const int lane = threadIdx.x & 31;
  const int wid  = (blockIdx.x * blockDim.x + threadIdx.x) >> 5;
  const int nStrips = N >> 6;
  const int mTiles  = M >> 5;
  if (wid >= mTiles * nStrips) return;      // wave-uniform: EXEC stays all-ones
  const int mt = wid / nStrips, ns = wid % nStrips;
  const int m0 = mt << 5, n0 = ns << 6;
  const int half = lane >> 4, r = lane & 15;

  v8f acc[2][4];
#pragma unroll
  for (int i = 0; i < 2; ++i)
#pragma unroll
    for (int j = 0; j < 4; ++j) acc[i][j] = (v8f){0,0,0,0,0,0,0,0};

  const unsigned short* a0p = A + (size_t)(m0 + r)      * K + half * 16;
  const unsigned short* a1p = A + (size_t)(m0 + 16 + r) * K + half * 16;
  const unsigned short* bp[4];
#pragma unroll
  for (int j = 0; j < 4; ++j)
    bp[j] = Bt + (size_t)(n0 + j * 16 + r) * K + half * 16;

  for (int k = 0; k < K; k += 32) {
    v16bf a0 = *(const v16bf*)(a0p + k);
    v16bf a1 = *(const v16bf*)(a1p + k);
#pragma unroll
    for (int j = 0; j < 4; ++j) {
      v16bf b = *(const v16bf*)(bp[j] + k);
      acc[0][j] = __builtin_amdgcn_wmma_f32_16x16x32_bf16(false, a0, false, b,
                                                          (short)0, acc[0][j], false, false);
      acc[1][j] = __builtin_amdgcn_wmma_f32_16x16x32_bf16(false, a1, false, b,
                                                          (short)0, acc[1][j], false, false);
    }
  }
  // C/D layout: VGPR v, lanes 0-15 -> row v, lanes 16-31 -> row v+8; col = lane%16
#pragma unroll
  for (int mi = 0; mi < 2; ++mi)
#pragma unroll
    for (int j = 0; j < 4; ++j)
#pragma unroll
      for (int v = 0; v < 8; ++v)
        C[(size_t)(m0 + mi * 16 + half * 8 + v) * N + n0 + j * 16 + r] = acc[mi][j][v];
}

// ---------------- self-loop + bias init: out = dinv^2 * h + bias ----------------
__global__ void __launch_bounds__(256) selfloop_init_kernel(const float* __restrict__ h,
                                                            const float* __restrict__ dinv,
                                                            const float* __restrict__ bias,
                                                            float* __restrict__ out,
                                                            long long total, int F) {
  long long i = (long long)blockIdx.x * blockDim.x + threadIdx.x;
  if (i < total) {
    int node = (int)(i / F);
    int f    = (int)(i % F);
    float di = dinv[node];
    out[i] = di * di * h[i] + bias[f];
  }
}

// ---------------- edge scatter, F=256: one wave per edge, lane covers 8 floats ----------------
__global__ void __launch_bounds__(256) edge_scatter256(const int* __restrict__ ei,
                                                       const float* __restrict__ h,
                                                       const float* __restrict__ dinv,
                                                       float* __restrict__ agg) {
  int lane = threadIdx.x & 31;
  long long w      = ((long long)blockIdx.x * blockDim.x + threadIdx.x) >> 5;
  long long stride = ((long long)gridDim.x * blockDim.x) >> 5;
  for (long long e = w; e < EE; e += stride) {
    int s = ei[e], d = ei[EE + e];
    float nrm = dinv[s] * dinv[d];
    const float4* hp = (const float4*)(h + (size_t)s * HID) + lane * 2;
    float*        op = agg + (size_t)d * HID + lane * 8;
    float4 v0 = hp[0], v1 = hp[1];
    atomicAdd(op + 0, v0.x * nrm); atomicAdd(op + 1, v0.y * nrm);
    atomicAdd(op + 2, v0.z * nrm); atomicAdd(op + 3, v0.w * nrm);
    atomicAdd(op + 4, v1.x * nrm); atomicAdd(op + 5, v1.y * nrm);
    atomicAdd(op + 6, v1.z * nrm); atomicAdd(op + 7, v1.w * nrm);
  }
}

// ---------------- edge scatter, F=128: one wave per edge, lane covers 4 floats ----------------
__global__ void __launch_bounds__(256) edge_scatter128(const int* __restrict__ ei,
                                                       const float* __restrict__ h,
                                                       const float* __restrict__ dinv,
                                                       float* __restrict__ agg) {
  int lane = threadIdx.x & 31;
  long long w      = ((long long)blockIdx.x * blockDim.x + threadIdx.x) >> 5;
  long long stride = ((long long)gridDim.x * blockDim.x) >> 5;
  for (long long e = w; e < EE; e += stride) {
    int s = ei[e], d = ei[EE + e];
    float nrm = dinv[s] * dinv[d];
    float4 v = *((const float4*)(h + (size_t)s * DOUT) + lane);
    float* op = agg + (size_t)d * DOUT + lane * 4;
    atomicAdd(op + 0, v.x * nrm); atomicAdd(op + 1, v.y * nrm);
    atomicAdd(op + 2, v.z * nrm); atomicAdd(op + 3, v.w * nrm);
  }
}

// ---------------- relu + bf16 convert (feeds GEMM2) ----------------
__global__ void __launch_bounds__(256) relu_cvt_kernel(const float* __restrict__ in,
                                                       unsigned short* __restrict__ out,
                                                       long long n) {
  long long i = (long long)blockIdx.x * blockDim.x + threadIdx.x;
  if (i < n) out[i] = f2bf(fmaxf(in[i], 0.0f));
}

extern "C" void kernel_launch(void* const* d_in, const int* in_sizes, int n_in,
                              void* d_out, int out_size, void* d_ws, size_t ws_size,
                              hipStream_t stream) {
  const float* x  = (const float*)d_in[0];
  const int*   ei = (const int*)d_in[1];     // [2,E] flat: [0..E) src, [E..2E) dst
  const float* W1 = (const float*)d_in[2];
  const float* b1 = (const float*)d_in[3];
  const float* W2 = (const float*)d_in[4];
  const float* b2 = (const float*)d_in[5];
  float* out = (float*)d_out;

  // workspace layout
  char* ws = (char*)d_ws;
  const size_t SZ_A    = (size_t)NN * FIN * 2;               // 102,400,000 B
  const size_t SZ_B    = (size_t)NN * HID * 4;               // 102,400,000 B
  const size_t OFF_W1B = SZ_A + SZ_B;
  const size_t OFF_W2B = OFF_W1B + (size_t)FIN * HID * 2;
  const size_t OFF_DI  = OFF_W2B + (size_t)HID * DOUT * 2;

  unsigned short* xb   = (unsigned short*)(ws + 0);          // bf16 x    [N,FIN]
  float*          agg1 = (float*)(ws + 0);                   // reuse A:  [N,HID] f32
  float*          h2   = (float*)(ws + 0);                   // reuse A:  [N,DOUT] f32
  float*          h1   = (float*)(ws + SZ_A);                // [N,HID] f32
  unsigned short* h1b  = (unsigned short*)(ws + SZ_A);       // reuse B: bf16 relu(h1)
  unsigned short* W1b  = (unsigned short*)(ws + OFF_W1B);    // bf16 W1^T [HID,FIN]
  unsigned short* W2b  = (unsigned short*)(ws + OFF_W2B);    // bf16 W2^T [DOUT,HID]
  float*          dinv = (float*)(ws + OFF_DI);              // [N]

  const int TB = 256;
  // 1) conversions
  {
    long long n = (long long)NN * FIN;
    cvt_bf16_kernel<<<(unsigned)((n + TB - 1) / TB), TB, 0, stream>>>(x, xb, n);
  }
  cvt_transpose_kernel<<<(FIN * HID + TB - 1) / TB, TB, 0, stream>>>(W1, W1b, FIN, HID);
  cvt_transpose_kernel<<<(HID * DOUT + TB - 1) / TB, TB, 0, stream>>>(W2, W2b, HID, DOUT);

  // 2) dst-degree (+ self-loop) -> dinv = rsqrt(deg)
  deg_init_kernel<<<(NN + TB - 1) / TB, TB, 0, stream>>>(dinv, NN);
  deg_accum_kernel<<<(EE + TB - 1) / TB, TB, 0, stream>>>(ei, dinv, EE);
  deg_rsqrt_kernel<<<(NN + TB - 1) / TB, TB, 0, stream>>>(dinv, NN);

  // 3) GEMM1: h1 = x @ W1   (M=100000, N=256, K=512); 32x64 tile/wave
  {
    int waves = (NN / 32) * (HID / 64);                      // 12500
    wmma_gemm_bf16<<<(waves + 3) / 4, 128, 0, stream>>>(xb, W1b, h1, NN, HID, FIN);
  }

  // 4) layer-1 aggregation: agg1 = dinv^2*h1 + b1, then edge scatter-adds
  {
    long long n = (long long)NN * HID;
    selfloop_init_kernel<<<(unsigned)((n + TB - 1) / TB), TB, 0, stream>>>(h1, dinv, b1, agg1, n, HID);
  }
  edge_scatter256<<<8192, TB, 0, stream>>>(ei, h1, dinv, agg1);

  // 5) relu + bf16
  {
    long long n = (long long)NN * HID;
    relu_cvt_kernel<<<(unsigned)((n + TB - 1) / TB), TB, 0, stream>>>(agg1, h1b, n);
  }

  // 6) GEMM2: h2 = relu(h1_agg) @ W2  (M=100000, N=128, K=256)
  {
    int waves = (NN / 32) * (DOUT / 64);                     // 6250
    wmma_gemm_bf16<<<(waves + 3) / 4, 128, 0, stream>>>(h1b, W2b, h2, NN, DOUT, HID);
  }

  // 7) layer-2 aggregation directly into d_out
  {
    long long n = (long long)NN * DOUT;
    selfloop_init_kernel<<<(unsigned)((n + TB - 1) / TB), TB, 0, stream>>>(h2, dinv, b2, out, n, DOUT);
  }
  edge_scatter128<<<8192, TB, 0, stream>>>(ei, h2, dinv, out);
}
